// CausalSelfAttention_8022998909588
// MI455X (gfx1250) — matrix operand order repaired and checked by reading, not compiled, not run on hardware
//
#include <hip/hip_runtime.h>

#define D_MODEL 768
#define N_HEAD  12
#define D_HEAD  64
#define BATCH   4
#define SEQ     2048
#define ROWS    (BATCH * SEQ)   // 8192

typedef __bf16 bf16;
typedef __attribute__((ext_vector_type(4)))  __bf16 v4bf;
typedef __attribute__((ext_vector_type(8)))  __bf16 v8bf;
typedef __attribute__((ext_vector_type(16))) __bf16 v16bf;
typedef __attribute__((ext_vector_type(8)))  float  v8f;
typedef __attribute__((ext_vector_type(4)))  int    v4i;

union Frag { v16bf v; v8bf h[2]; };

#define WMMA_BF16(a, b, c) \
  __builtin_amdgcn_wmma_f32_16x16x32_bf16(false, (a), false, (b), (short)0, (c), false, false)

#if defined(__gfx1250__) && __has_builtin(__builtin_amdgcn_global_load_async_to_lds_b128)
#define HAVE_ASYNC_LDS 1
#else
#define HAVE_ASYNC_LDS 0
#endif

// copy 16 bytes global -> LDS (async path when available)
__device__ __forceinline__ void copy16_g2l(const bf16* g, bf16* l) {
#if HAVE_ASYNC_LDS
  __builtin_amdgcn_global_load_async_to_lds_b128(
      (__attribute__((address_space(1))) v4i*)g,
      (__attribute__((address_space(3))) v4i*)l, 0, 0);
#else
  *(v8bf*)l = *(const v8bf*)g;
#endif
}

__device__ __forceinline__ void wait_async_copies() {
#if HAVE_ASYNC_LDS
#if __has_builtin(__builtin_amdgcn_s_wait_asynccnt)
  __builtin_amdgcn_s_wait_asynccnt(0);
#else
  asm volatile("s_wait_asynccnt 0" ::: "memory");
#endif
#endif
}

// A-fragment (16x32, row per lane): lane m<16 holds K[k0+half*8 .. +8) and
// K[k0+16+half*8 .. +8) of row m (ISA 7.12.2 16-bit A layout).
__device__ __forceinline__ v16bf load_afrag(const bf16* rowbase, int k0, int half) {
  Frag f;
  f.h[0] = *(const v8bf*)(rowbase + k0 + half * 8);
  f.h[1] = *(const v8bf*)(rowbase + k0 + 16 + half * 8);
  return f.v;
}

// B-fragment (32x16, column per lane): lane n&15 holds K[k0+half*8 .. +8) and
// K[k0+16+half*8 .. +8) of column n (same k order as the A fragment).
__device__ __forceinline__ v16bf load_bfrag(const bf16* colbase, int k0, int half) {
  Frag f;
  f.h[0] = *(const v8bf*)(colbase + k0 + half * 8);
  f.h[1] = *(const v8bf*)(colbase + k0 + 16 + half * 8);
  return f.v;
}

__device__ __forceinline__ float hmax16(float v) {
  #pragma unroll
  for (int m = 1; m < 16; m <<= 1) v = fmaxf(v, __shfl_xor(v, m, 16));
  return v;
}
__device__ __forceinline__ float hsum16(float v) {
  #pragma unroll
  for (int m = 1; m < 16; m <<= 1) v += __shfl_xor(v, m, 16);
  return v;
}

// ---------------- fp32 -> bf16 conversion (4 elements / thread) ----------------
__global__ __launch_bounds__(256) void k_cvt(const float* __restrict__ s,
                                             bf16* __restrict__ d, int n) {
  int i = (blockIdx.x * blockDim.x + threadIdx.x) * 4;
  if (i < n) {
    float4 f = *(const float4*)(s + i);
    v4bf o = { (bf16)f.x, (bf16)f.y, (bf16)f.z, (bf16)f.w };
    *(v4bf*)(d + i) = o;
  }
}

// ---------------- V -> V^T per head: vt[b][h][d][t] = v[b][t][h*64+d] ----------
__global__ __launch_bounds__(256) void k_transpose_v(const bf16* __restrict__ v,
                                                     bf16* __restrict__ vt) {
  int idx = blockIdx.x * blockDim.x + threadIdx.x;   // output index, t fastest
  int t    = idx & (SEQ - 1);
  int rest = idx >> 11;          // (b*H + h)*64 + d
  int d    = rest & 63;
  int bh   = rest >> 6;
  int b = bh / N_HEAD, h = bh % N_HEAD;
  vt[idx] = v[(size_t)(b * SEQ + t) * D_MODEL + h * D_HEAD + d];
}

// ---------------- WMMA GEMM: C[M,N] = A[M,K] * Bw[N,K]^T + bias ----------------
// Block tile 128(M) x 64(N), 8 waves each computing a 32x32 sub-tile.
// A/B K-slices staged in LDS (async global->LDS when available), rows padded to
// 40 bf16 (80B) so fragment reads are LDS-bank-conflict-free.
#define LROW 40
__global__ __launch_bounds__(256) void k_gemm(const bf16* __restrict__ A,
                                              const bf16* __restrict__ Bw,
                                              const float* __restrict__ bias,
                                              bf16* __restrict__ Cb,
                                              float* __restrict__ Cf,
                                              int M, int N, int K) {
  __shared__ bf16 ldsA[128 * LROW];   // 10.0 KB
  __shared__ bf16 ldsB[64 * LROW];    //  5.0 KB

  const int nb_n = N >> 6;                         // blocks along N
  const int bm = blockIdx.x / nb_n, bn = blockIdx.x % nb_n;
  const int m0 = bm * 128, n0 = bn * 64;

  const int t    = threadIdx.x;
  const int lane = t & 31;
  const int w    = t >> 5;                         // wave in block: 0..7
  const int wm   = w >> 1, wn = w & 1;             // 4x2 wave grid
  const int half = lane >> 4, l16 = lane & 15;

  v8f acc[2][2] = {};

  for (int k0 = 0; k0 < K; k0 += 32) {
    __syncthreads();                               // previous tile fully read
    // stage A: 128 rows x 32 k = 512 16B-chunks; 2 per thread
    #pragma unroll
    for (int j = 0; j < 2; ++j) {
      int task = t + j * 256;
      int row = task >> 2, ch = task & 3;
      copy16_g2l(A + (size_t)(m0 + row) * K + k0 + ch * 8,
                 &ldsA[row * LROW + ch * 8]);
    }
    // stage B: 64 rows x 32 k = 256 16B-chunks; 1 per thread
    {
      int row = t >> 2, ch = t & 3;
      copy16_g2l(Bw + (size_t)(n0 + row) * K + k0 + ch * 8,
                 &ldsB[row * LROW + ch * 8]);
    }
    wait_async_copies();
    __syncthreads();                               // tile visible to all waves

    Frag a0, a1, b0, b1;
    a0.h[0] = *(const v8bf*)&ldsA[(wm * 32 + l16) * LROW + half * 8];
    a0.h[1] = *(const v8bf*)&ldsA[(wm * 32 + l16) * LROW + 16 + half * 8];
    a1.h[0] = *(const v8bf*)&ldsA[(wm * 32 + 16 + l16) * LROW + half * 8];
    a1.h[1] = *(const v8bf*)&ldsA[(wm * 32 + 16 + l16) * LROW + 16 + half * 8];
    b0.h[0] = *(const v8bf*)&ldsB[(wn * 32 + l16) * LROW + half * 8];
    b0.h[1] = *(const v8bf*)&ldsB[(wn * 32 + l16) * LROW + 16 + half * 8];
    b1.h[0] = *(const v8bf*)&ldsB[(wn * 32 + 16 + l16) * LROW + half * 8];
    b1.h[1] = *(const v8bf*)&ldsB[(wn * 32 + 16 + l16) * LROW + 16 + half * 8];

    acc[0][0] = WMMA_BF16(a0.v, b0.v, acc[0][0]);
    acc[0][1] = WMMA_BF16(a0.v, b1.v, acc[0][1]);
    acc[1][0] = WMMA_BF16(a1.v, b0.v, acc[1][0]);
    acc[1][1] = WMMA_BF16(a1.v, b1.v, acc[1][1]);
  }

  #pragma unroll
  for (int ti = 0; ti < 2; ++ti) {
    #pragma unroll
    for (int tj = 0; tj < 2; ++tj) {
      const int n = n0 + wn * 32 + tj * 16 + l16;  // C layout: N = lane&15
      const float bv = bias[n];
      #pragma unroll
      for (int i = 0; i < 8; ++i) {                // C layout: M = vgpr + 8*half
        int m = m0 + wm * 32 + ti * 16 + i + 8 * half;
        float val = acc[ti][tj][i] + bv;
        if (Cb) Cb[(size_t)m * N + n] = (bf16)val;
        if (Cf) Cf[(size_t)m * N + n] = val;
      }
    }
  }
}

// ---------------- fused causal flash attention -------------------------------
// One wave owns a 16-row Q tile of one (b,h); loops KV in 32-wide tiles.
__global__ __launch_bounds__(256) void k_flash(const bf16* __restrict__ Q,
                                               const bf16* __restrict__ Kb,
                                               const bf16* __restrict__ VT,
                                               bf16* __restrict__ Y) {
  __shared__ bf16 ldsP[8][16 * 32];        // wave-private 16x32 P staging
  const int wib   = threadIdx.x >> 5;
  const int lane  = threadIdx.x & 31;
  const int gwave = blockIdx.x * 8 + wib;
  const int nqt   = SEQ / 16;              // 128 q tiles per (b,h)
  const int bh    = gwave / nqt;
  const int qt    = gwave % nqt;
  if (bh >= BATCH * N_HEAD) return;
  const int b = bh / N_HEAD, h = bh % N_HEAD;
  const int q0 = qt * 16;
  const int half = lane >> 4, l16 = lane & 15;

  // Q A-fragments: 16 rows x 64 features = 2 fragments (K=32 each)
  const bf16* qrow = Q + (size_t)(b * SEQ + q0 + l16) * D_MODEL + h * D_HEAD;
  const v16bf qf0 = load_afrag(qrow, 0,  half);
  const v16bf qf1 = load_afrag(qrow, 32, half);

  v8f o0 = {}, o1 = {}, o2 = {}, o3 = {};  // 16 rows x 64 feats (4 C tiles)
  float mrow[8], lrow[8];
  #pragma unroll
  for (int i = 0; i < 8; ++i) { mrow[i] = -1e30f; lrow[i] = 0.f; }

  const bf16* kbase  = Kb + (size_t)(b * SEQ) * D_MODEL + h * D_HEAD;
  const bf16* vtbase = VT + (size_t)bh * D_HEAD * SEQ;
  const float scale = 0.125f;              // 1/sqrt(64)

  for (int kv0 = 0; kv0 < q0 + 16; kv0 += 32) {
    // ---- S = Q * K^T for 32 kv columns (2 N-halves x 2 feature chunks) ----
    const bf16* kcol0 = kbase + (size_t)(kv0 + l16) * D_MODEL;
    const bf16* kcol1 = kbase + (size_t)(kv0 + 16 + l16) * D_MODEL;
    v8f s0 = {}, s1 = {};
    s0 = WMMA_BF16(qf0, load_bfrag(kcol0, 0,  half), s0);
    s0 = WMMA_BF16(qf1, load_bfrag(kcol0, 32, half), s0);
    s1 = WMMA_BF16(qf0, load_bfrag(kcol1, 0,  half), s1);
    s1 = WMMA_BF16(qf1, load_bfrag(kcol1, 32, half), s1);

    // ---- scale + causal mask + online softmax (rows live per lane-half) ----
    float p0[8], p1[8];
    #pragma unroll
    for (int i = 0; i < 8; ++i) {
      const int row = q0 + i + 8 * half;
      float a = s0[i] * scale; if (kv0 + l16      > row) a = -1e30f;
      float c = s1[i] * scale; if (kv0 + 16 + l16 > row) c = -1e30f;
      float tmax  = hmax16(fmaxf(a, c));
      float mnew  = fmaxf(mrow[i], tmax);
      float alpha = __expf(mrow[i] - mnew);
      float e0 = __expf(a - mnew);
      float e1 = __expf(c - mnew);
      float rs = hsum16(e0 + e1);
      lrow[i] = lrow[i] * alpha + rs;
      mrow[i] = mnew;
      o0[i] *= alpha; o1[i] *= alpha; o2[i] *= alpha; o3[i] *= alpha;
      p0[i] = e0; p1[i] = e1;
    }

    // ---- re-shape P (C layout) -> A fragment via wave-private LDS ----
    bf16* pp = &ldsP[wib][0];
    #pragma unroll
    for (int i = 0; i < 8; ++i) {
      pp[(i + 8 * half) * 32 + l16]      = (bf16)p0[i];
      pp[(i + 8 * half) * 32 + 16 + l16] = (bf16)p1[i];
    }
    __builtin_amdgcn_wave_barrier();       // keep ds store->load order
    Frag pf;
    pf.h[0] = *(const v8bf*)&pp[l16 * 32 + half * 8];
    pf.h[1] = *(const v8bf*)&pp[l16 * 32 + 16 + half * 8];
    __builtin_amdgcn_wave_barrier();

    // ---- O += P * V (V^T columns are contiguous along t) ----
    o0 = WMMA_BF16(pf.v, load_bfrag(vtbase + (size_t)(l16)      * SEQ, kv0, half), o0);
    o1 = WMMA_BF16(pf.v, load_bfrag(vtbase + (size_t)(16 + l16) * SEQ, kv0, half), o1);
    o2 = WMMA_BF16(pf.v, load_bfrag(vtbase + (size_t)(32 + l16) * SEQ, kv0, half), o2);
    o3 = WMMA_BF16(pf.v, load_bfrag(vtbase + (size_t)(48 + l16) * SEQ, kv0, half), o3);
  }

  // ---- normalize and store y (bf16, feeds final projection GEMM) ----
  bf16* ybase = Y + (size_t)(b * SEQ + q0) * D_MODEL + h * D_HEAD;
  #pragma unroll
  for (int i = 0; i < 8; ++i) {
    const float inv = 1.0f / lrow[i];
    bf16* yr = ybase + (size_t)(i + 8 * half) * D_MODEL;
    yr[l16]      = (bf16)(o0[i] * inv);
    yr[16 + l16] = (bf16)(o1[i] * inv);
    yr[32 + l16] = (bf16)(o2[i] * inv);
    yr[48 + l16] = (bf16)(o3[i] * inv);
  }
}

extern "C" void kernel_launch(void* const* d_in, const int* in_sizes, int n_in,
                              void* d_out, int out_size, void* d_ws, size_t ws_size,
                              hipStream_t stream) {
  const float* x  = (const float*)d_in[0];
  const float* Wq = (const float*)d_in[1];
  const float* bq = (const float*)d_in[2];
  const float* Wk = (const float*)d_in[3];
  const float* bk = (const float*)d_in[4];
  const float* Wv = (const float*)d_in[5];
  const float* bv = (const float*)d_in[6];
  const float* Wp = (const float*)d_in[7];
  const float* bp = (const float*)d_in[8];
  float* out = (float*)d_out;

  const size_t XSZ = (size_t)ROWS * D_MODEL;      // 6.29M elems
  const size_t WSZ = (size_t)D_MODEL * D_MODEL;   // 0.59M elems
  bf16* p = (bf16*)d_ws;
  bf16* x_bf  = p; p += XSZ;
  bf16* wq_bf = p; p += WSZ;
  bf16* wk_bf = p; p += WSZ;
  bf16* wv_bf = p; p += WSZ;
  bf16* wp_bf = p; p += WSZ;
  bf16* q_bf  = p; p += XSZ;
  bf16* k_bf  = p; p += XSZ;
  bf16* v_bf  = p; p += XSZ;
  bf16* vt_bf = p; p += XSZ;
  bf16* y_bf  = p; p += XSZ;                      // total ~80 MB

  k_cvt<<<(int)(XSZ / 4 / 256), 256, 0, stream>>>(x,  x_bf,  (int)XSZ);
  k_cvt<<<(int)(WSZ / 4 / 256), 256, 0, stream>>>(Wq, wq_bf, (int)WSZ);
  k_cvt<<<(int)(WSZ / 4 / 256), 256, 0, stream>>>(Wk, wk_bf, (int)WSZ);
  k_cvt<<<(int)(WSZ / 4 / 256), 256, 0, stream>>>(Wv, wv_bf, (int)WSZ);
  k_cvt<<<(int)(WSZ / 4 / 256), 256, 0, stream>>>(Wp, wp_bf, (int)WSZ);

  const int gemm_blocks = (ROWS / 128) * (D_MODEL / 64);      // 64*12 = 768
  k_gemm<<<gemm_blocks, 256, 0, stream>>>(x_bf, wq_bf, bq, q_bf, nullptr,
                                          ROWS, D_MODEL, D_MODEL);
  k_gemm<<<gemm_blocks, 256, 0, stream>>>(x_bf, wk_bf, bk, k_bf, nullptr,
                                          ROWS, D_MODEL, D_MODEL);
  k_gemm<<<gemm_blocks, 256, 0, stream>>>(x_bf, wv_bf, bv, v_bf, nullptr,
                                          ROWS, D_MODEL, D_MODEL);

  k_transpose_v<<<(int)(XSZ / 256), 256, 0, stream>>>(v_bf, vt_bf);

  const int att_blocks = BATCH * N_HEAD * (SEQ / 16) / 8;     // 768
  k_flash<<<att_blocks, 256, 0, stream>>>(q_bf, k_bf, vt_bf, y_bf);

  k_gemm<<<gemm_blocks, 256, 0, stream>>>(y_bf, wp_bf, bp, nullptr, out,
                                          ROWS, D_MODEL, D_MODEL);
}
